// EnhancedBiLSTM_CRF_84756884619425
// MI455X (gfx1250) — compile-verified, matrix operand order checked by reading, not run on hardware
//
#include <hip/hip_runtime.h>
#include <math.h>

// ---------------------------------------------------------------------------
// EnhancedBiLSTM-CRF forward for MI455X (gfx1250, wave32).
// - Every GEMM-shaped op -> v_wmma_f32_16x16x32_bf16 (bf16 in, f32 acc).
// - Big GEMMs: 8-wave workgroup, 32x64 macro tile; A chunk staged to LDS with
//   global_load_async_to_lds_b64 (ASYNCcnt) and shared by all 8 waves;
//   B fragments as 2x global_load_b128 per lane; W prefetch via
//   global_prefetch_b8.
// - Sequential BiLSTM recurrence: 128 per-step launches per layer, WMMA
//   matvec ([64x192]x[768x192]^T both dirs) + fused gate kernel.
// - Pointwise (LN, gelu, softmax, highway, LSTM cell, CRF): wave32 VALU.
// ---------------------------------------------------------------------------

typedef __attribute__((ext_vector_type(16))) __bf16 v16bf;
typedef __attribute__((ext_vector_type(8)))  __bf16 v8bf;
typedef __attribute__((ext_vector_type(8)))  float  v8f;

#define Bn 64
#define Sn 128
#define Tn 8192   // Bn*Sn tokens
#define HDn 384
#define NTAG 9

__device__ __forceinline__ __bf16 f2bf(float f) {
  union { float f; unsigned u; } v; v.f = f;
  unsigned r = v.u + 0x7FFFu + ((v.u >> 16) & 1u);
  unsigned short h = (unsigned short)(r >> 16);
  __bf16 o; __builtin_memcpy(&o, &h, sizeof(o)); return o;
}
__device__ __forceinline__ float bf2f(__bf16 b) {
  unsigned short h; __builtin_memcpy(&h, &b, sizeof(h));
  union { unsigned u; float f; } v; v.u = ((unsigned)h) << 16; return v.f;
}
__device__ __forceinline__ float sigf(float x) { return 1.f / (1.f + expf(-x)); }
__device__ __forceinline__ float act_f(float v, int act) {
  if (act == 1) return v > 0.f ? v : 0.f;                                   // relu
  if (act == 2) return 0.5f * v * (1.f + erff(v * 0.70710678118654752f));   // exact gelu
  if (act == 3) return sigf(v);                                             // sigmoid
  return v;
}

// Load a 16x32 bf16 A/B fragment from row-major memory (ld elements, mult of 8).
// Lane L: row = tm + (L&15); K elements per ISA 7.12.2 layout; two b128 loads.
__device__ __forceinline__ v16bf frag_load(const __bf16* __restrict__ p, int ld,
                                           int row, int k0, int lane) {
  const int r = lane & 15, kh = lane >> 4;
  const __bf16* q = p + (size_t)(row + r) * ld + k0 + (kh << 3);
  v8bf lo = *(const v8bf*)q;
  v8bf hi = *(const v8bf*)(q + 16);
  return __builtin_shufflevector(lo, hi, 0,1,2,3,4,5,6,7,8,9,10,11,12,13,14,15);
}
#define WMMA_BF16(a, b, c) \
  __builtin_amdgcn_wmma_f32_16x16x32_bf16(false, (a), false, (b), (short)0, (c), false, false)

// ------------------------------ prep kernels -------------------------------

// f32 [N][K] -> bf16 [Npad][Kpad], zero padded
__global__ void k_padw(const float* __restrict__ src, __bf16* __restrict__ dst,
                       int N, int K, int Npad, int Kpad) {
  int i = blockIdx.x * 256 + threadIdx.x;
  if (i >= Npad * Kpad) return;
  int r = i / Kpad, c = i % Kpad;
  float v = (r < N && c < K) ? src[(size_t)r * K + c] : 0.f;
  dst[i] = f2bf(v);
}

// conv w [100][100][ksz] -> bf16 slices [sbase+kk][112][128]
__global__ void k_pad_convw(const float* __restrict__ src, __bf16* __restrict__ dst,
                            int ksz, int sbase) {
  int i = blockIdx.x * 256 + threadIdx.x;
  if (i >= ksz * 112 * 128) return;
  int kk = i / (112 * 128); int rem = i % (112 * 128);
  int ch = rem / 128, cd = rem % 128;
  float v = (ch < 100 && cd < 100) ? src[(size_t)(ch * 100 + cd) * ksz + kk] : 0.f;
  dst[(size_t)(sbase + kk) * (112 * 128) + rem] = f2bf(v);
}

__global__ void k_addvec(const float* __restrict__ a, const float* __restrict__ b,
                         float* __restrict__ o, int n) {
  int i = blockIdx.x * 256 + threadIdx.x;
  if (i < n) o[i] = a[i] + b[i];
}

// ----------------------------- macro-tile GEMM -----------------------------
// C[M x N] = act(A[M x K]bf16 * W[N x K]bf16^T + bias); M mult of 32.
// 256 threads = 8 waves; macro tile 32(M) x 64(N); wave grid 2x4.
// Per K-chunk: 2KB A tile staged to LDS via global_load_async_to_lds_b64
// (one b64 per lane, ASYNCcnt), shared across all 8 waves.
__global__ void k_gemm(const __bf16* __restrict__ A, int lda,
                       const __bf16* __restrict__ W, int ldw,
                       const float* __restrict__ bias,
                       float* __restrict__ C, int ldc,
                       __bf16* __restrict__ Cb, int ldcb,
                       int M, int N, int K, int act) {
  __shared__ __align__(16) __bf16 sA[32 * 32];
  const int tid = threadIdx.x;
  const int lane = tid & 31;
  const int wave = tid >> 5;               // 0..7
  const int wm = wave >> 2, wn = wave & 3; // 2 x 4 wave grid
  const int nmac = (N + 63) >> 6;
  const int tmmac = (blockIdx.x / nmac) << 5;
  const int tn = ((blockIdx.x % nmac) << 6) + (wn << 4);
  // staging: thread -> (row = tid/8, 4 bf16 at col (tid%8)*4) of the 32x32 chunk
  const int srow = tid >> 3;
  const int scol = (tid & 7) << 2;
  const __bf16* Ag = A + (size_t)(tmmac + srow) * lda + scol;
  const unsigned ldsa = (unsigned)(uintptr_t)&sA[srow * 32 + scol];

  v8f acc = {};
  for (int k = 0; k < K; k += 32) {
    __syncthreads();                       // previous chunk fully consumed
    asm volatile("global_load_async_to_lds_b64 %0, %1, off"
                 :: "v"(ldsa), "v"(Ag + k) : "memory");
    if (k + 32 < K)
      __builtin_prefetch((const void*)(W + (size_t)(tn + (lane & 15)) * ldw + k + 32), 0, 1);
    asm volatile("s_wait_asynccnt 0x0" ::: "memory");
    __syncthreads();                       // all waves' fills visible
    const __bf16* qa = &sA[((wm << 4) + (lane & 15)) * 32 + ((lane >> 4) << 3)];
    v8bf lo = *(const v8bf*)qa;
    v8bf hi = *(const v8bf*)(qa + 16);
    v16bf a = __builtin_shufflevector(lo, hi, 0,1,2,3,4,5,6,7,8,9,10,11,12,13,14,15);
    v16bf b = frag_load(W, ldw, tn, k, lane);
    acc = WMMA_BF16(a, b, acc);
  }
  const int n = tn + (lane & 15);
  if (n >= N) return;
  const float bv = bias ? bias[n] : 0.f;
  const int mh = (lane >> 4) << 3;
  const int tm = tmmac + (wm << 4);
#pragma unroll
  for (int r = 0; r < 8; ++r) {
    int m = tm + mh + r;
    float v = act_f(acc[r] + bv, act);
    if (C)  C[(size_t)m * ldc + n] = v;
    if (Cb) Cb[(size_t)m * ldcb + n] = f2bf(v);
  }
}

// ----------------------------- char CNN path -------------------------------

// ce_pad[t][p+3][cd] = char_emb[char_ids[t][p]][cd]; [Tn][32][128] bf16 (pre-zeroed)
__global__ void k_char_embed(const int* __restrict__ char_ids,
                             const float* __restrict__ char_emb,
                             __bf16* __restrict__ ce) {
  int i = blockIdx.x * 256 + threadIdx.x;
  if (i >= Tn * 25 * 100) return;
  int c = i % 100; int r = i / 100; int p = r % 25; int t = r / 25;
  int id = char_ids[t * 25 + p];
  ce[((size_t)t * 32 + (p + 3)) * 128 + c] = f2bf(char_emb[(size_t)id * 100 + c]);
}

// conv + relu + max over positions; one wave per 16-token x 16-channel tile.
__global__ void k_conv(const __bf16* __restrict__ ce, const __bf16* __restrict__ wc,
                       const float* __restrict__ bias, __bf16* __restrict__ cfb,
                       int ksz, int pad, int sbase, int colbase) {
  const int lane = threadIdx.x & 31;
  const int tm = (blockIdx.x / 7) << 4;
  const int tn = (blockIdx.x % 7) << 4;
  const int n = tn + (lane & 15);
  const float bv = (n < 100) ? bias[n] : 0.f;
  float best[8];
#pragma unroll
  for (int r = 0; r < 8; ++r) best[r] = 0.f;   // relu>=0, so 0 is a valid identity
  for (int p = 0; p < 25; ++p) {
    v8f acc = {};
    for (int kk = 0; kk < ksz; ++kk) {
      const __bf16* ab = ce + (size_t)tm * 4096 + (size_t)(p + kk + 3 - pad) * 128;
      const __bf16* wb = wc + (size_t)(sbase + kk) * (112 * 128);
      for (int c = 0; c < 128; c += 32) {
        v16bf a = frag_load(ab, 4096, 0, c, lane);
        v16bf b = frag_load(wb, 128, tn, c, lane);
        acc = WMMA_BF16(a, b, acc);
      }
    }
#pragma unroll
    for (int r = 0; r < 8; ++r) {
      float v = acc[r] + bv; v = v > 0.f ? v : 0.f;
      best[r] = fmaxf(best[r], v);
    }
  }
  if (n < 100) {
    const int mh = (lane >> 4) << 3;
#pragma unroll
    for (int r = 0; r < 8; ++r)
      cfb[(size_t)(tm + mh + r) * 320 + colbase + n] = f2bf(best[r]);
  }
}

// highway combine in place on cfb: cf = g*relu_nl + (1-g)*cf (g,nl precomputed)
__global__ void k_highway(const float* __restrict__ nl, const float* __restrict__ gt,
                          __bf16* __restrict__ cfb) {
  int i = blockIdx.x * 256 + threadIdx.x;
  if (i >= Tn * 300) return;
  int c = i % 300, t = i / 300;
  float g = gt[(size_t)t * 304 + c];
  float cf = bf2f(cfb[(size_t)t * 320 + c]);
  cfb[(size_t)t * 320 + c] = f2bf(g * nl[(size_t)t * 304 + c] + (1.f - g) * cf);
}

// x_in[t][0..299]=word_emb, [300..399]=char_proj, [400..415]=0  (bf16, ld 416)
__global__ void k_build_x(const int* __restrict__ word_ids, const float* __restrict__ word_emb,
                          const float* __restrict__ cp, __bf16* __restrict__ x) {
  int i = blockIdx.x * 256 + threadIdx.x;
  if (i >= Tn * 416) return;
  int c = i % 416, t = i / 416;
  float v = 0.f;
  if (c < 300)       v = word_emb[(size_t)word_ids[t] * 300 + c];
  else if (c < 400)  v = cp[(size_t)t * 112 + (c - 300)];
  x[i] = f2bf(v);
}

// ------------------------------- layernorm ---------------------------------
// one wave per 384-wide row: t = X[+R]; LN; [gelu]; [+R]; emit f32/bf16
__global__ void k_ln(const float* __restrict__ X, const float* __restrict__ R,
                     const float* __restrict__ g, const float* __restrict__ b,
                     float* __restrict__ fout, __bf16* __restrict__ bout,
                     int preadd, int dogelu, int postadd) {
  int row = blockIdx.x * 8 + (threadIdx.x >> 5);
  int lane = threadIdx.x & 31;
  const float* x = X + (size_t)row * HDn;
  const float* rr = R ? R + (size_t)row * HDn : nullptr;
  float v[12]; float s = 0.f, sq = 0.f;
#pragma unroll
  for (int j = 0; j < 12; ++j) {
    int c = lane + 32 * j;
    float t = x[c];
    if (preadd) t += rr[c];
    v[j] = t; s += t; sq += t * t;
  }
  for (int o = 16; o; o >>= 1) { s += __shfl_xor(s, o, 32); sq += __shfl_xor(sq, o, 32); }
  float mean = s * (1.f / HDn);
  float var = sq * (1.f / HDn) - mean * mean;
  float rstd = rsqrtf(var + 1e-5f);
#pragma unroll
  for (int j = 0; j < 12; ++j) {
    int c = lane + 32 * j;
    float t = (v[j] - mean) * rstd * g[c] + b[c];
    if (dogelu)  t = 0.5f * t * (1.f + erff(t * 0.70710678118654752f));
    if (postadd) t += rr[c];
    if (fout) fout[(size_t)row * HDn + c] = t;
    if (bout) bout[(size_t)row * HDn + c] = f2bf(t);
  }
}

// --------------------------------- BiLSTM ----------------------------------
// gadd[d][64][768] = h_prev[d] @ whh[d]^T  (WMMA; grid (4*48, 2), 32 thr)
__global__ void k_lstm_matvec(const __bf16* __restrict__ hprev,
                              const __bf16* __restrict__ whh0,
                              const __bf16* __restrict__ whh1,
                              float* __restrict__ gadd) {
  const int lane = threadIdx.x & 31;
  const int d = blockIdx.y;
  const int tm = (blockIdx.x / 48) << 4;
  const int tn = (blockIdx.x % 48) << 4;
  const __bf16* A = hprev + (size_t)d * 64 * 192;
  const __bf16* W = d ? whh1 : whh0;
  v8f acc = {};
  for (int k = 0; k < 192; k += 32)
    acc = WMMA_BF16(frag_load(A, 192, tm, k, lane), frag_load(W, 192, tn, k, lane), acc);
  const int n = tn + (lane & 15), mh = (lane >> 4) << 3;
  float* C = gadd + (size_t)d * 64 * 768;
#pragma unroll
  for (int r = 0; r < 8; ++r) C[(size_t)(tm + mh + r) * 768 + n] = acc[r];
}

// fused LSTM cell for both directions; i over 2*64*192 units
__global__ void k_lstm_cell(const float* __restrict__ pre0, const float* __restrict__ pre1,
                            const float* __restrict__ gadd,
                            __bf16* __restrict__ hprev, float* __restrict__ cstate,
                            __bf16* __restrict__ hsb, float* __restrict__ hsf, int t) {
  int i = blockIdx.x * 256 + threadIdx.x;
  if (i >= 2 * 64 * 192) return;
  int d = i / (64 * 192); int rem = i % (64 * 192);
  int b = rem / 192, u = rem % 192;
  int td = d ? (Sn - 1 - t) : t;
  const float* pre = d ? pre1 : pre0;
  size_t rp = (size_t)(b * Sn + td) * 768;
  const float* ga = gadd + (size_t)d * 64 * 768 + (size_t)b * 768;
  float gi = pre[rp + u]       + ga[u];
  float gf = pre[rp + 192 + u] + ga[192 + u];
  float gg = pre[rp + 384 + u] + ga[384 + u];
  float go = pre[rp + 576 + u] + ga[576 + u];
  float c = cstate[i];
  c = sigf(gf) * c + sigf(gi) * tanhf(gg);
  float h = sigf(go) * tanhf(c);
  cstate[i] = c;
  hprev[i] = f2bf(h);
  size_t orow = (size_t)(b * Sn + td) * HDn + d * 192 + u;
  hsb[orow] = f2bf(h);
  hsf[orow] = h;
}

// -------------------------------- attention --------------------------------

__global__ void k_qkv_split(const float* __restrict__ qkv, __bf16* __restrict__ qb,
                            __bf16* __restrict__ kb, __bf16* __restrict__ vt) {
  int i = blockIdx.x * 256 + threadIdx.x;
  if (i >= Bn * Sn * 4 * 96) return;
  int d = i % 96; int r = i / 96; int h = r % 4; r /= 4; int s = r % Sn; int b = r / Sn;
  const float* src = qkv + (size_t)(b * Sn + s) * 1152 + h * 96 + d;
  size_t qi = ((size_t)(b * 4 + h) * Sn + s) * 96 + d;
  qb[qi] = f2bf(src[0]);
  kb[qi] = f2bf(src[384]);
  vt[((size_t)(b * 4 + h) * 96 + d) * Sn + s] = f2bf(src[768]);
}

__global__ void k_attn_scores(const __bf16* __restrict__ qb, const __bf16* __restrict__ kb,
                              float* __restrict__ sc) {
  const int lane = threadIdx.x & 31;
  const int bh = blockIdx.y;
  const int tm = (blockIdx.x >> 3) << 4;
  const int tn = (blockIdx.x & 7) << 4;
  const __bf16* Q = qb + (size_t)bh * Sn * 96;
  const __bf16* Kx = kb + (size_t)bh * Sn * 96;
  v8f acc = {};
  for (int k = 0; k < 96; k += 32)
    acc = WMMA_BF16(frag_load(Q, 96, tm, k, lane), frag_load(Kx, 96, tn, k, lane), acc);
  const int n = tn + (lane & 15), mh = (lane >> 4) << 3;
  float* out = sc + (size_t)bh * Sn * Sn;
#pragma unroll
  for (int r = 0; r < 8; ++r)
    out[(size_t)(tm + mh + r) * Sn + n] = acc[r] * 0.10206207261596575f; // 1/sqrt(96)
}

__global__ void k_softmax(const float* __restrict__ sc, __bf16* __restrict__ pb) {
  int row = blockIdx.x * 8 + (threadIdx.x >> 5);   // Bn*4*Sn rows
  int lane = threadIdx.x & 31;
  const float* r = sc + (size_t)row * Sn;
  float v[4]; float mx = -1e30f;
#pragma unroll
  for (int j = 0; j < 4; ++j) { v[j] = r[lane + 32 * j]; mx = fmaxf(mx, v[j]); }
  for (int o = 16; o; o >>= 1) mx = fmaxf(mx, __shfl_xor(mx, o, 32));
  float s = 0.f;
#pragma unroll
  for (int j = 0; j < 4; ++j) { v[j] = expf(v[j] - mx); s += v[j]; }
  for (int o = 16; o; o >>= 1) s += __shfl_xor(s, o, 32);
  float inv = 1.f / s;
#pragma unroll
  for (int j = 0; j < 4; ++j) pb[(size_t)row * Sn + lane + 32 * j] = f2bf(v[j] * inv);
}

__global__ void k_attn_av(const __bf16* __restrict__ pb, const __bf16* __restrict__ vt,
                          __bf16* __restrict__ ob) {
  const int lane = threadIdx.x & 31;
  const int bh = blockIdx.y; const int b = bh >> 2; const int h = bh & 3;
  const int tm = (blockIdx.x / 6) << 4;
  const int tn = (blockIdx.x % 6) << 4;
  const __bf16* P = pb + (size_t)bh * Sn * Sn;
  const __bf16* V = vt + (size_t)bh * 96 * Sn;
  v8f acc = {};
  for (int k = 0; k < Sn; k += 32)
    acc = WMMA_BF16(frag_load(P, Sn, tm, k, lane), frag_load(V, Sn, tn, k, lane), acc);
  const int n = tn + (lane & 15), mh = (lane >> 4) << 3;
#pragma unroll
  for (int r = 0; r < 8; ++r) {
    int q = tm + mh + r;
    ob[(size_t)(b * Sn + q) * HDn + h * 96 + n] = f2bf(acc[r]);
  }
}

// ---------------------------------- CRF ------------------------------------
__global__ void k_crf(const float* __restrict__ em, const int* __restrict__ labels,
                      const int* __restrict__ mask, const float* __restrict__ trans,
                      float* __restrict__ out) {
  __shared__ float sh[Bn];
  int b = threadIdx.x;
  const float* eb = em + (size_t)b * Sn * 16;
  const int* lb = labels + b * Sn;
  const int* mb = mask + b * Sn;
  float score = 0.f; int prevtag = 0;
  for (int t = 0; t < Sn; ++t) {
    int lab = lb[t]; int tag = (lab == -100) ? 0 : lab;
    float mf = (float)mb[t];
    score += eb[t * 16 + tag] * mf;
    if (t > 0) score += trans[prevtag * NTAG + tag] * mf;
    prevtag = tag;
  }
  float la[NTAG];
#pragma unroll
  for (int j = 0; j < NTAG; ++j) la[j] = eb[j];
  for (int t = 1; t < Sn; ++t) {
    float mf = (float)mb[t];
    float nl[NTAG];
#pragma unroll
    for (int j = 0; j < NTAG; ++j) {
      float mx = -1e30f;
      for (int i = 0; i < NTAG; ++i) mx = fmaxf(mx, la[i] + trans[i * NTAG + j]);
      float ss = 0.f;
      for (int i = 0; i < NTAG; ++i) ss += expf(la[i] + trans[i * NTAG + j] - mx);
      nl[j] = mx + logf(ss) + eb[t * 16 + j];
    }
#pragma unroll
    for (int j = 0; j < NTAG; ++j) la[j] = (mf > 0.f) ? nl[j] : la[j];
  }
  float mx = -1e30f;
#pragma unroll
  for (int j = 0; j < NTAG; ++j) mx = fmaxf(mx, la[j]);
  float ss = 0.f;
#pragma unroll
  for (int j = 0; j < NTAG; ++j) ss += expf(la[j] - mx);
  sh[b] = (mx + logf(ss)) - score;
  __syncthreads();
  if (b == 0) {
    float s = 0.f;
    for (int k = 0; k < Bn; ++k) s += sh[k];
    out[0] = s * (1.f / Bn);
  }
}

// --------------------------------- driver ----------------------------------

extern "C" void kernel_launch(void* const* d_in, const int* in_sizes, int n_in,
                              void* d_out, int out_size, void* d_ws, size_t ws_size,
                              hipStream_t stream) {
  (void)in_sizes; (void)n_in; (void)out_size; (void)ws_size;
  // inputs (setup_inputs insertion order; params nested in insertion order)
  const int*   word_ids = (const int*)d_in[0];
  const int*   char_ids = (const int*)d_in[1];
  const int*   labels   = (const int*)d_in[2];
  const int*   maskp    = (const int*)d_in[3];
  const float* word_emb = (const float*)d_in[4];
  const float* char_emb = (const float*)d_in[5];
  auto F = [&](int i) { return (const float*)d_in[i]; };
  // 6..11 conv{1,2,3}.{w,b}; 12..17 hw_nl/hw_gate/char_proj; 18,19 comb;
  // 20,21 comb_ln; 22..45 lstm[l][d]{wih,whh,bih,bhh}; 46,47 lstm_proj;
  // 48,49 lstm_ln; 50,51 qkv; 52,53 attn_out; 54,55 ff1; 56,57 ff2;
  // 58..63 norm1/norm2/normf; 64,65 head1; 66,67 head2; 68 trans.

  char* base = (char*)d_ws;
  size_t off = 0;
  auto alloc = [&](size_t bytes) -> void* {
    void* p = (void*)(base + off);
    off = off + ((bytes + 255) & ~(size_t)255);
    return p;
  };

  // ---- bf16 weight buffers (N padded to 64 for the macro-tile GEMM) ----
  __bf16* wconvb = (__bf16*)alloc((size_t)15 * 112 * 128 * 2);
  __bf16* whwnl  = (__bf16*)alloc((size_t)320 * 320 * 2);
  __bf16* whwgt  = (__bf16*)alloc((size_t)320 * 320 * 2);
  __bf16* wcp    = (__bf16*)alloc((size_t)128 * 320 * 2);
  __bf16* wcomb  = (__bf16*)alloc((size_t)384 * 416 * 2);
  __bf16 *wihb[6], *whhb[6];
  for (int i = 0; i < 6; ++i) {
    wihb[i] = (__bf16*)alloc((size_t)768 * 384 * 2);
    whhb[i] = (__bf16*)alloc((size_t)768 * 192 * 2);
  }
  __bf16* wproj = (__bf16*)alloc((size_t)384 * 384 * 2);
  __bf16* wqkv  = (__bf16*)alloc((size_t)1152 * 384 * 2);
  __bf16* wattn = (__bf16*)alloc((size_t)384 * 384 * 2);
  __bf16* wff1  = (__bf16*)alloc((size_t)1536 * 384 * 2);
  __bf16* wff2  = (__bf16*)alloc((size_t)384 * 1536 * 2);
  __bf16* whd1  = (__bf16*)alloc((size_t)384 * 384 * 2);
  __bf16* whd2  = (__bf16*)alloc((size_t)64 * 384 * 2);
  float*  bsumd = (float*)alloc((size_t)6 * 768 * 4);

  // ---- activations ----
  __bf16* cepad = (__bf16*)alloc((size_t)Tn * 32 * 128 * 2);
  __bf16* cfb   = (__bf16*)alloc((size_t)Tn * 320 * 2);
  float*  nlb   = (float*)alloc((size_t)Tn * 304 * 4);
  float*  gtb   = (float*)alloc((size_t)Tn * 304 * 4);
  float*  cpb   = (float*)alloc((size_t)Tn * 112 * 4);
  __bf16* x_in  = (__bf16*)alloc((size_t)Tn * 416 * 2);
  float*  combo = (float*)alloc((size_t)Tn * HDn * 4);
  __bf16* x0b   = (__bf16*)alloc((size_t)Tn * HDn * 2);
  float*  pre0  = (float*)alloc((size_t)Tn * 768 * 4);
  float*  pre1  = (float*)alloc((size_t)Tn * 768 * 4);
  __bf16* hprevb= (__bf16*)alloc((size_t)2 * 64 * 192 * 2);
  float*  cstate= (float*)alloc((size_t)2 * 64 * 192 * 4);
  float*  gadd  = (float*)alloc((size_t)2 * 64 * 768 * 4);
  __bf16* hsb   = (__bf16*)alloc((size_t)Tn * HDn * 2);
  float*  hsf   = (float*)alloc((size_t)Tn * HDn * 4);
  float*  projo = (float*)alloc((size_t)Tn * HDn * 4);
  float*  hbuf  = (float*)alloc((size_t)Tn * HDn * 4);
  __bf16* hb    = (__bf16*)alloc((size_t)Tn * HDn * 2);
  float*  qkvf  = (float*)alloc((size_t)Tn * 1152 * 4);
  __bf16* qb    = (__bf16*)alloc((size_t)Bn * 4 * Sn * 96 * 2);
  __bf16* kb    = (__bf16*)alloc((size_t)Bn * 4 * Sn * 96 * 2);
  __bf16* vt    = (__bf16*)alloc((size_t)Bn * 4 * 96 * Sn * 2);
  float*  scb   = (float*)alloc((size_t)Bn * 4 * Sn * Sn * 4);
  __bf16* pbb   = (__bf16*)alloc((size_t)Bn * 4 * Sn * Sn * 2);
  __bf16* obb   = (__bf16*)alloc((size_t)Tn * HDn * 2);
  float*  aout  = (float*)alloc((size_t)Tn * HDn * 4);
  float*  h1f   = (float*)alloc((size_t)Tn * HDn * 4);
  __bf16* h1b   = (__bf16*)alloc((size_t)Tn * HDn * 2);
  __bf16* ffb   = (__bf16*)alloc((size_t)Tn * 1536 * 2);
  float*  ff2o  = (float*)alloc((size_t)Tn * HDn * 4);
  float*  h2f   = (float*)alloc((size_t)Tn * HDn * 4);
  __bf16* hfb   = (__bf16*)alloc((size_t)Tn * HDn * 2);
  __bf16* hd1b  = (__bf16*)alloc((size_t)Tn * HDn * 2);
  float*  emis  = (float*)alloc((size_t)Tn * 16 * 4);

  auto padw = [&](const float* src, __bf16* dst, int N, int K, int Np, int Kp) {
    int n = Np * Kp;
    k_padw<<<(n + 255) / 256, 256, 0, stream>>>(src, dst, N, K, Np, Kp);
  };
  auto gemm = [&](const __bf16* A, int lda, const __bf16* W, int ldw, const float* bias,
                  float* C, int ldc, __bf16* Cb, int ldcb, int M, int N, int K, int act) {
    int blocks = (M / 32) * ((N + 63) / 64);   // 32x64 macro tiles, 8 waves each
    k_gemm<<<blocks, 256, 0, stream>>>(A, lda, W, ldw, bias, C, ldc, Cb, ldcb, M, N, K, act);
  };

  // ---- weight conversion ----
  k_pad_convw<<<(3 * 112 * 128 + 255) / 256, 256, 0, stream>>>(F(6),  wconvb, 3, 0);
  k_pad_convw<<<(5 * 112 * 128 + 255) / 256, 256, 0, stream>>>(F(8),  wconvb, 5, 3);
  k_pad_convw<<<(7 * 112 * 128 + 255) / 256, 256, 0, stream>>>(F(10), wconvb, 7, 8);
  padw(F(12), whwnl, 300, 300, 320, 320);
  padw(F(14), whwgt, 300, 300, 320, 320);
  padw(F(16), wcp, 100, 300, 128, 320);
  padw(F(18), wcomb, 384, 400, 384, 416);
  for (int l = 0; l < 3; ++l)
    for (int d = 0; d < 2; ++d) {
      int ib = 22 + l * 8 + d * 4, i = l * 2 + d;
      padw(F(ib + 0), wihb[i], 768, 384, 768, 384);
      padw(F(ib + 1), whhb[i], 768, 192, 768, 192);
      k_addvec<<<3, 256, 0, stream>>>(F(ib + 2), F(ib + 3), bsumd + i * 768, 768);
    }
  padw(F(46), wproj, 384, 384, 384, 384);
  padw(F(50), wqkv, 1152, 384, 1152, 384);
  padw(F(52), wattn, 384, 384, 384, 384);
  padw(F(54), wff1, 1536, 384, 1536, 384);
  padw(F(56), wff2, 384, 1536, 384, 1536);
  padw(F(64), whd1, 384, 384, 384, 384);
  padw(F(66), whd2, 9, 384, 64, 384);

  // ---- char CNN + highway + projection ----
  hipMemsetAsync(cepad, 0, (size_t)Tn * 32 * 128 * 2, stream);
  hipMemsetAsync(cfb, 0, (size_t)Tn * 320 * 2, stream);
  k_char_embed<<<(Tn * 25 * 100 + 255) / 256, 256, 0, stream>>>(char_ids, char_emb, cepad);
  k_conv<<<512 * 7, 32, 0, stream>>>(cepad, wconvb, F(7),  cfb, 3, 1, 0, 0);
  k_conv<<<512 * 7, 32, 0, stream>>>(cepad, wconvb, F(9),  cfb, 5, 2, 3, 100);
  k_conv<<<512 * 7, 32, 0, stream>>>(cepad, wconvb, F(11), cfb, 7, 3, 8, 200);
  gemm(cfb, 320, whwnl, 320, F(13), nlb, 304, nullptr, 0, Tn, 300, 320, 1); // relu
  gemm(cfb, 320, whwgt, 320, F(15), gtb, 304, nullptr, 0, Tn, 300, 320, 3); // sigmoid
  k_highway<<<(Tn * 300 + 255) / 256, 256, 0, stream>>>(nlb, gtb, cfb);
  gemm(cfb, 320, wcp, 320, F(17), cpb, 112, nullptr, 0, Tn, 100, 320, 0);
  k_build_x<<<(Tn * 416 + 255) / 256, 256, 0, stream>>>(word_ids, word_emb, cpb, x_in);
  gemm(x_in, 416, wcomb, 416, F(19), combo, HDn, nullptr, 0, Tn, 384, 416, 0);
  k_ln<<<Tn / 8, 256, 0, stream>>>(combo, nullptr, F(20), F(21), nullptr, x0b, 0, 1, 0);

  // ---- 3-layer BiLSTM ----
  for (int l = 0; l < 3; ++l) {
    const __bf16* inA = (l == 0) ? x0b : hsb;
    gemm(inA, HDn, wihb[2 * l],     384, bsumd + (2 * l) * 768,     pre0, 768, nullptr, 0, Tn, 768, 384, 0);
    gemm(inA, HDn, wihb[2 * l + 1], 384, bsumd + (2 * l + 1) * 768, pre1, 768, nullptr, 0, Tn, 768, 384, 0);
    hipMemsetAsync(hprevb, 0, (size_t)2 * 64 * 192 * 2, stream);
    hipMemsetAsync(cstate, 0, (size_t)2 * 64 * 192 * 4, stream);
    for (int t = 0; t < Sn; ++t) {
      k_lstm_matvec<<<dim3(4 * 48, 2), 32, 0, stream>>>(hprevb, whhb[2 * l], whhb[2 * l + 1], gadd);
      k_lstm_cell<<<(2 * 64 * 192 + 255) / 256, 256, 0, stream>>>(pre0, pre1, gadd, hprevb,
                                                                  cstate, hsb, hsf, t);
    }
  }

  // ---- projection + residual, transformer block ----
  gemm(hsb, HDn, wproj, 384, F(47), projo, HDn, nullptr, 0, Tn, 384, 384, 0);
  k_ln<<<Tn / 8, 256, 0, stream>>>(projo, hsf, F(48), F(49), hbuf, hb, 0, 1, 1);
  gemm(hb, HDn, wqkv, 384, F(51), qkvf, 1152, nullptr, 0, Tn, 1152, 384, 0);
  k_qkv_split<<<(Bn * Sn * 4 * 96 + 255) / 256, 256, 0, stream>>>(qkvf, qb, kb, vt);
  k_attn_scores<<<dim3(64, Bn * 4), 32, 0, stream>>>(qb, kb, scb);
  k_softmax<<<(Bn * 4 * Sn) / 8, 256, 0, stream>>>(scb, pbb);
  k_attn_av<<<dim3(8 * 6, Bn * 4), 32, 0, stream>>>(pbb, vt, obb);
  gemm(obb, HDn, wattn, 384, F(53), aout, HDn, nullptr, 0, Tn, 384, 384, 0);
  k_ln<<<Tn / 8, 256, 0, stream>>>(aout, hbuf, F(58), F(59), h1f, h1b, 1, 0, 0);   // norm1
  gemm(h1b, HDn, wff1, 384, F(55), nullptr, 0, ffb, 1536, Tn, 1536, 384, 2);       // gelu
  gemm(ffb, 1536, wff2, 1536, F(57), ff2o, HDn, nullptr, 0, Tn, 384, 1536, 0);
  k_ln<<<Tn / 8, 256, 0, stream>>>(ff2o, h1f, F(60), F(61), h2f, nullptr, 1, 0, 0); // norm2
  k_ln<<<Tn / 8, 256, 0, stream>>>(h2f, nullptr, F(62), F(63), nullptr, hfb, 0, 0, 0); // normf

  // ---- heads + CRF ----
  gemm(hfb, HDn, whd1, 384, F(65), nullptr, 0, hd1b, HDn, Tn, 384, 384, 2); // gelu
  gemm(hd1b, HDn, whd2, 384, F(67), emis, 16, nullptr, 0, Tn, 9, 384, 0);
  k_crf<<<1, Bn, 0, stream>>>(emis, labels, maskp, F(68), (float*)d_out);
}
